// FirSourceNet_11570641896039
// MI455X (gfx1250) — matrix operand order repaired and checked by reading, not compiled, not run on hardware
//
#include <hip/hip_runtime.h>
#include <hip/hip_bf16.h>

// ---------------------------------------------------------------------------
// FirSourceNet on MI455X (gfx1250, wave32, WMMA).
// GEMM-heavy parts (pointwise convs, DFT-as-GEMM STFT/iSTFT) use
// v_wmma_f32_16x16x32_f16 with f16 operands staged through LDS via the
// CDNA5 async-to-LDS DMA path (ASYNCcnt), double-buffered.
// ---------------------------------------------------------------------------

typedef _Float16 v16h __attribute__((ext_vector_type(16)));
typedef _Float16 v8h  __attribute__((ext_vector_type(8)));
typedef float    v8f  __attribute__((ext_vector_type(8)));

#define EPI_F32      0   // outf = acc (+bias)
#define EPI_GELU_F16 1   // outh = f16(gelu(acc + bias))
#define EPI_RES_F32  2   // outf = acc + bias + res
#define EPI_ELU1_F32 3   // outf = elu(acc+bias)+1 for n<Nvalid else 0
#define EPI_F16      4   // outh = f16(acc)

static __device__ __forceinline__ float gelu_exact(float v) {
    return 0.5f * v * (1.0f + erff(v * 0.70710678118654752f));
}

// 32-bit LDS offset of a shared-memory pointer (generic LDS addr = {aperture, off})
static __device__ __forceinline__ unsigned lds_off32(const void* p) {
    return (unsigned)(size_t)p;
}

// ---------------------------------------------------------------------------
// Generic f16 GEMM:  C[M][Nn] = A[M][K] * W[Nn][K]^T   (row-major, lda=ldb=K)
// block = 256 threads = 8 waves; tile 64(M) x 128(N); wave tile 16 x 64.
// Staging: global_load_async_to_lds_b128 into double-buffered LDS.
// ---------------------------------------------------------------------------
template <int EPI>
__global__ __launch_bounds__(256)
void gemm_wmma_f16(const _Float16* __restrict__ A, const _Float16* __restrict__ W,
                   const float* __restrict__ bias, const float* __restrict__ res,
                   float* __restrict__ outf, _Float16* __restrict__ outh,
                   int M, int Nn, int K, int Nvalid)
{
    __shared__ _Float16 As[2][64][40];    // +8 halves pad: rows stay 16B aligned
    __shared__ _Float16 Bs[2][128][40];

    const int tid  = threadIdx.x;
    const int lane = tid & 31;
    const int wave = tid >> 5;
    const int m0   = blockIdx.y * 64;
    const int n0   = blockIdx.x * 128;
    const int mw   = m0 + 16 * (wave & 3);   // wave M origin
    const int nw   = n0 + 64 * (wave >> 2);  // wave N origin

    v8f acc[4];
#pragma unroll
    for (int j = 0; j < 4; ++j)
#pragma unroll
        for (int r = 0; r < 8; ++r) acc[j][r] = 0.0f;

    // staging maps: A tile 64x32 (one b128/lane), B tile 128x32 (two b128/lane)
    const int arow = tid >> 2, acol = (tid & 3) * 8;
    const int brow = tid >> 1, bcol = (tid & 1) * 16;
    // clamp rows: DMA cannot zero-fill; garbage rows only feed unstored outputs
    const int gm = min(m0 + arow, M - 1);
    const int gn = min(n0 + brow, Nn - 1);
    const _Float16* gA = A + (size_t)gm * K + acol;
    const _Float16* gB = W + (size_t)gn * K + bcol;

    const int steps = K >> 5;

    auto issue_async = [&](int kk, int buf) {
        const int k0 = kk << 5;
        unsigned la = lds_off32(&As[buf][arow][acol]);
        unsigned lb = lds_off32(&Bs[buf][brow][bcol]);
        const _Float16* pa = gA + k0;
        const _Float16* pb = gB + k0;
        asm volatile("global_load_async_to_lds_b128 %0, %1, off"
                     :: "v"(la), "v"(pa) : "memory");
        asm volatile("global_load_async_to_lds_b128 %0, %1, off"
                     :: "v"(lb), "v"(pb) : "memory");
        asm volatile("global_load_async_to_lds_b128 %0, %1, off offset:16"
                     :: "v"(lb + 16u), "v"(pb) : "memory");
    };

    issue_async(0, 0);

    const int mrow = (mw - m0) + (lane & 15);
    const int ka0  = (lane >> 4) * 8;
    const int kb0  = (lane >> 4) * 8;

    for (int kk = 0; kk < steps; ++kk) {
        const int cur = kk & 1;
        // this wave's fills for buf[cur] done; barrier makes them WG-visible and
        // guarantees every wave finished reading buf[cur^1] (prev iteration)
        asm volatile("s_wait_asynccnt 0x0" ::: "memory");
        __syncthreads();
        if (kk + 1 < steps) {
            issue_async(kk + 1, cur ^ 1);            // overlaps WMMA below
            __builtin_prefetch(gB + ((kk + 1) << 5), 0, 1);  // global_prefetch_b8
        }

        // A fragment: lane -> row m, two 8-K chunks (VGPR layout per ISA 7.12.2)
        v8h a_lo = *(const v8h*)&As[cur][mrow][ka0];
        v8h a_hi = *(const v8h*)&As[cur][mrow][ka0 + 16];
        v16h afrag;
#pragma unroll
        for (int i = 0; i < 8; ++i) { afrag[i] = a_lo[i]; afrag[8 + i] = a_hi[i]; }

        v16h bfrag[4];
#pragma unroll
        for (int j = 0; j < 4; ++j) {
            const int nrow = (nw - n0) + 16 * j + (lane & 15);
            v8h b_lo = *(const v8h*)&Bs[cur][nrow][kb0];
            v8h b_hi = *(const v8h*)&Bs[cur][nrow][kb0 + 16];
#pragma unroll
            for (int i = 0; i < 8; ++i) { bfrag[j][i] = b_lo[i]; bfrag[j][8 + i] = b_hi[i]; }
        }
#pragma unroll
        for (int j = 0; j < 4; ++j)
            acc[j] = __builtin_amdgcn_wmma_f32_16x16x32_f16(
                         false, afrag, false, bfrag[j], (short)0, acc[j], false, false);
    }

    // epilogue; C/D layout: lane n = lane&15, VGPR r -> m = r + 8*(lane>>4)
    const int cn = lane & 15;
    const int mr = (lane >> 4) * 8;
#pragma unroll
    for (int j = 0; j < 4; ++j) {
        const int n = nw + 16 * j + cn;
#pragma unroll
        for (int r = 0; r < 8; ++r) {
            const int m = mw + mr + r;
            if (m >= M || n >= Nn) continue;
            float v = acc[j][r];
            const size_t o = (size_t)m * Nn + n;
            if (EPI == EPI_F32) {
                if (bias) v += bias[n];
                outf[o] = v;
            } else if (EPI == EPI_GELU_F16) {
                v += bias[n];
                outh[o] = (_Float16)gelu_exact(v);
            } else if (EPI == EPI_RES_F32) {
                v += bias[n] + res[o];
                outf[o] = v;
            } else if (EPI == EPI_ELU1_F32) {
                float y = 0.0f;
                if (n < Nvalid) { v += bias[n]; y = (v > 0.0f) ? v + 1.0f : __expf(v); }
                outf[o] = y;
            } else {
                outh[o] = (_Float16)v;
            }
        }
    }
}

// ---------------------------------------------------------------------------
// small helpers
// ---------------------------------------------------------------------------
__global__ void cvt_f32_f16_pad(const float* __restrict__ in, _Float16* __restrict__ out,
                                int rows_out, int rows_in, int cols)
{
    long idx = (long)blockIdx.x * 256 + threadIdx.x;
    long tot = (long)rows_out * cols;
    if (idx >= tot) return;
    int r = (int)(idx / cols);
    out[idx] = (r < rows_in) ? (_Float16)in[idx] : (_Float16)0.0f;
}

__global__ void xpose_x_f16(const float* __restrict__ x, _Float16* __restrict__ xh)
{
    // x:(8,192,500) -> xh[m=n*500+l][c]
    int idx = blockIdx.x * 256 + threadIdx.x;
    if (idx >= 4000 * 192) return;
    int c = idx % 192, m = idx / 192;
    int n = m / 500, l = m % 500;
    xh[idx] = (_Float16)x[((size_t)n * 192 + c) * 500 + l];
}

__global__ void prep_scalars(const float* __restrict__ f0, const float* __restrict__ energy,
                             float* __restrict__ lf0, float* __restrict__ en)
{
    int m = blockIdx.x * 256 + threadIdx.x;
    if (m >= 4000) return;
    int n = m / 500, l = m % 500;
    float f = f0[n * 500 + l];
    lf0[m] = logf(fmaxf(f, 0.0f) + 1e-6f);
    en[m]  = energy[n * 500 + l];
}

__global__ void spk_proj(const float* __restrict__ spk, const float* __restrict__ sw,
                         const float* __restrict__ cb, const float* __restrict__ sb,
                         const float* __restrict__ fb, const float* __restrict__ eb,
                         float* __restrict__ outb)
{
    int n = blockIdx.x, o = threadIdx.x;   // grid 8, block 512
    float s = 0.0f;
    for (int c = 0; c < 256; ++c) s += sw[o * 256 + c] * spk[n * 256 + c];
    outb[n * 512 + o] = s + cb[o] + sb[o] + fb[o] + eb[o];
}

// LayerNorm over 512 channels for one position per block (256 thr = 8 waves)
__global__ __launch_bounds__(256)
void layernorm_c(const float* __restrict__ in, float* __restrict__ outf,
                 _Float16* __restrict__ outh,
                 const float* __restrict__ g, const float* __restrict__ b,
                 const float* __restrict__ spkbias, const float* __restrict__ fw,
                 const float* __restrict__ ew, const float* __restrict__ lf0,
                 const float* __restrict__ en)
{
    __shared__ float red[16];
    int m = blockIdx.x;
    int n = m / 500;
    int tid = threadIdx.x;
    float v0 = in[(size_t)m * 512 + tid];
    float v1 = in[(size_t)m * 512 + tid + 256];
    if (spkbias) {
        v0 += spkbias[n * 512 + tid]       + fw[tid]       * lf0[m] + ew[tid]       * en[m];
        v1 += spkbias[n * 512 + tid + 256] + fw[tid + 256] * lf0[m] + ew[tid + 256] * en[m];
    }
    float s = v0 + v1, s2 = v0 * v0 + v1 * v1;
    for (int off = 16; off; off >>= 1) { s += __shfl_xor(s, off, 32); s2 += __shfl_xor(s2, off, 32); }
    int wave = tid >> 5, lane = tid & 31;
    if (lane == 0) { red[wave] = s; red[8 + wave] = s2; }
    __syncthreads();
    if (tid == 0) {
        float ts = 0.0f, t2 = 0.0f;
        for (int i = 0; i < 8; ++i) { ts += red[i]; t2 += red[8 + i]; }
        red[0] = ts; red[8] = t2;
    }
    __syncthreads();
    float mean = red[0] * (1.0f / 512.0f);
    float var  = red[8] * (1.0f / 512.0f) - mean * mean;
    float rs   = rsqrtf(var + 1e-6f);
    float o0 = (v0 - mean) * rs * g[tid]       + b[tid];
    float o1 = (v1 - mean) * rs * g[tid + 256] + b[tid + 256];
    if (outf) { outf[(size_t)m * 512 + tid] = o0; outf[(size_t)m * 512 + tid + 256] = o1; }
    if (outh) { outh[(size_t)m * 512 + tid] = (_Float16)o0; outh[(size_t)m * 512 + tid + 256] = (_Float16)o1; }
}

__global__ void dwconv7(const float* __restrict__ in, float* __restrict__ out,
                        const float* __restrict__ w, const float* __restrict__ b)
{
    int idx = blockIdx.x * 256 + threadIdx.x;
    if (idx >= 4000 * 512) return;
    int c = idx & 511;
    int m = idx >> 9;
    int n = m / 500, l = m % 500;
    float acc = b[c];
#pragma unroll
    for (int j = 0; j < 7; ++j) {
        int ll = l + j - 3;
        if (ll >= 0 && ll < 500)
            acc += w[c * 7 + j] * in[((size_t)(n * 500 + ll)) * 512 + c];
    }
    out[(size_t)m * 512 + c] = acc;
}

__global__ void amps_head(const float* __restrict__ h, const float* __restrict__ aw,
                          const float* __restrict__ ab, float* __restrict__ amp)
{
    int wave = threadIdx.x >> 5, lane = threadIdx.x & 31;
    int m = blockIdx.x * 8 + wave;
    if (m >= 4000) return;
    float s = 0.0f;
    for (int c = lane; c < 512; c += 32) s += aw[c] * h[(size_t)m * 512 + c];
    for (int off = 16; off; off >>= 1) s += __shfl_xor(s, off, 32);
    if (lane == 0) {
        float v = s + ab[0];
        amp[m] = (v > 0.0f) ? v + 1.0f : __expf(v);     // elu(v)+1
    }
}

// ---------------------------------------------------------------------------
// oscillator (phase cumsum: f32 in-frame, f64 frame hierarchy)
// ---------------------------------------------------------------------------
__device__ __forceinline__ float fs_at(const float* __restrict__ f0n, int s)
{
    float pos = (s + 0.5f) * (1.0f / 960.0f) - 0.5f;
    pos = fminf(fmaxf(pos, 0.0f), 499.0f);
    int lo = (int)pos;
    int hi = min(lo + 1, 499);
    float w = pos - (float)lo;
    return f0n[lo] * (1.0f - w) + f0n[hi] * w;
}

__global__ void frame_phase_sum(const float* __restrict__ f0, float* __restrict__ S)
{
    __shared__ float red[8];
    int l = blockIdx.x, n = blockIdx.y;
    const float* f0n = f0 + n * 500;
    float part = 0.0f;
    for (int t = threadIdx.x; t < 960; t += 256) part += fs_at(f0n, l * 960 + t);
    for (int off = 16; off; off >>= 1) part += __shfl_xor(part, off, 32);
    if ((threadIdx.x & 31) == 0) red[threadIdx.x >> 5] = part;
    __syncthreads();
    if (threadIdx.x == 0) {
        float t = 0.0f;
        for (int i = 0; i < 8; ++i) t += red[i];
        S[n * 500 + l] = t * (float)(M_PI / 48000.0);
    }
}

__global__ void scan_frames(const float* __restrict__ S, double* __restrict__ O)
{
    __shared__ double sh[512];
    int n = blockIdx.x, l = threadIdx.x;
    double v = (l < 500) ? (double)S[n * 500 + l] : 0.0;
    sh[l] = v;
    __syncthreads();
    for (int off = 1; off < 512; off <<= 1) {
        double t = (l >= off) ? sh[l - off] : 0.0;
        __syncthreads();
        sh[l] += t;
        __syncthreads();
    }
    if (l < 500) O[n * 500 + l] = sh[l] - v;   // exclusive prefix
}

__global__ void oscillator(const float* __restrict__ f0, const double* __restrict__ O,
                           const float* __restrict__ amp, float* __restrict__ harm)
{
    int l = blockIdx.x, n = blockIdx.y, lane = threadIdx.x;   // block = 32 (1 wave)
    const float* f0n  = f0 + n * 500;
    const float* ampn = amp + n * 500;
    int s0 = l * 960 + lane * 30;
    float part = 0.0f;
    for (int t = 0; t < 30; ++t) part += fs_at(f0n, s0 + t);
    part *= (float)(M_PI / 48000.0);
    float incl = part;
    for (int off = 1; off < 32; off <<= 1) {
        float v = __shfl_up(incl, off, 32);
        if (lane >= off) incl += v;
    }
    double ph = O[n * 500 + l] + (double)(incl - part);
    float uv = (f0n[l] > 20.0f) ? 1.0f : 0.0f;
    for (int t = 0; t < 30; ++t) {
        int s = s0 + t;
        float fs = fs_at(f0n, s);
        ph += (double)fs * (M_PI / 48000.0);
        float pix = (float)fmod(ph, M_PI);
        float a = rintf(48000.0f / fmaxf(fs, 20.0f) * 0.5f) * 2.0f + 1.0f;
        float osc = (pix < 1e-8f) ? 1.0f : sinf(a * pix) / (a * sinf(pix));
        float pos = (s + 0.5f) * (1.0f / 960.0f) - 0.5f;
        pos = fminf(fmaxf(pos, 0.0f), 499.0f);
        int lo = (int)pos;
        int hi = min(lo + 1, 499);
        float w = pos - (float)lo;
        float am = ampn[lo] * (1.0f - w) + ampn[hi] * w;
        harm[(size_t)n * 480000 + s] = osc * uv * am;
    }
}

// ---------------------------------------------------------------------------
// STFT noise path: framing, DFT basis tables, spectral multiply, overlap-add
// ---------------------------------------------------------------------------
__global__ void make_frames(const float* __restrict__ noise, _Float16* __restrict__ frames)
{
    int f = blockIdx.x, n = blockIdx.y;    // 501 frames x 8 batch
    const float* xn = noise + (size_t)n * 480000;
    _Float16* fr = frames + ((size_t)n * 501 + f) * 3840;
    for (int t = threadIdx.x; t < 3840; t += 256) {
        int i = f * 960 + t - 1920;        // reflect padding
        if (i < 0) i = -i;
        if (i >= 480000) i = 2 * 480000 - 2 - i;
        float wv = 0.5f - 0.5f * cosf((float)t * (float)(2.0 * M_PI / 3840.0));
        fr[t] = (_Float16)(xn[i] * wv);
    }
}

__global__ void build_ftab(_Float16* __restrict__ Ftab)
{
    int r = blockIdx.x;                    // 0..3871; rows 0..1935 = Re, rest = Im
    int k = (r < 1936) ? r : r - 1936;
    bool im = (r >= 1936);
    _Float16* row = Ftab + (size_t)r * 3840;
    for (int t = threadIdx.x; t < 3840; t += 256) {
        float v = 0.0f;
        if (k <= 1920) {
            int ph = (int)(((long long)k * t) % 3840);   // exact integer phase
            float ang = (float)ph * (float)(2.0 * M_PI / 3840.0);
            v = im ? -sinf(ang) : cosf(ang);
        }
        row[t] = (_Float16)v;
    }
}

__global__ void build_itab(_Float16* __restrict__ Itab)
{
    int t = blockIdx.x;                    // 0..3839; hann + 1/N folded in
    float wt = 0.5f - 0.5f * cosf((float)t * (float)(2.0 * M_PI / 3840.0));
    _Float16* row = Itab + (size_t)t * 3872;
    for (int c = threadIdx.x; c < 3872; c += 256) {
        int k = (c < 1936) ? c : c - 1936;
        bool im = (c >= 1936);
        float v = 0.0f;
        if (k <= 1920) {
            float cf = (k == 0 || k == 1920) ? 1.0f : 2.0f;
            int ph = (int)(((long long)k * t) % 3840);
            float ang = (float)ph * (float)(2.0 * M_PI / 3840.0);
            v = (im ? -sinf(ang) : cosf(ang)) * cf * wt * (1.0f / 3840.0f);
        }
        row[c] = (_Float16)v;
    }
}

__global__ void apply_kernels(_Float16* __restrict__ Spec, const float* __restrict__ Ker)
{
    long idx = (long)blockIdx.x * 256 + threadIdx.x;
    if (idx >= (long)4008 * 3872) return;
    int c = (int)(idx % 3872);
    int m = (int)(idx / 3872);
    int n = m / 501, g = m % 501;
    int bin = (c < 1936) ? c : c - 1936;
    float v = 0.0f;
    if (g > 0 && bin <= 1920)
        v = (float)Spec[idx] * Ker[((size_t)(n * 500 + g - 1)) * 1936 + bin];
    Spec[idx] = (_Float16)v;               // in-place, pure elementwise
}

__global__ void ola_final(const _Float16* __restrict__ yt, const float* __restrict__ harm,
                          float* __restrict__ out)
{
    int s = blockIdx.x * 256 + threadIdx.x;   // 1875*256 == 480000
    int n = blockIdx.y;
    int p = s + 1920;
    int gmin = (p - 3839 + 959) / 960;  if (gmin < 0)   gmin = 0;
    int gmax = p / 960;                 if (gmax > 500) gmax = 500;
    float acc = 0.0f, env = 0.0f;
    for (int g = gmin; g <= gmax; ++g) {
        int off = p - g * 960;
        float wt = 0.5f - 0.5f * cosf((float)off * (float)(2.0 * M_PI / 3840.0));
        env += wt * wt;
        acc += (float)yt[((size_t)n * 501 + g) * 3840 + off];
    }
    float y = acc / fmaxf(env, 1e-11f);
    float o = harm[(size_t)n * 480000 + s] + y;
    out[(size_t)n * 480000 + s] = o;
    out[(size_t)3840000 + (size_t)n * 480000 + s] = o;   // duplicated tuple output
}

// ---------------------------------------------------------------------------
// host launcher
// ---------------------------------------------------------------------------
extern "C" void kernel_launch(void* const* d_in, const int* in_sizes, int n_in,
                              void* d_out, int out_size, void* d_ws, size_t ws_size,
                              hipStream_t stream)
{
    (void)in_sizes; (void)n_in; (void)out_size; (void)ws_size;

    const float* x      = (const float*)d_in[0];
    const float* f0     = (const float*)d_in[1];
    const float* energy = (const float*)d_in[2];
    const float* spk    = (const float*)d_in[3];
    const float* noise  = (const float*)d_in[4];
    const float* cw   = (const float*)d_in[5];
    const float* cb   = (const float*)d_in[6];
    const float* sw   = (const float*)d_in[7];
    const float* sb   = (const float*)d_in[8];
    const float* ew   = (const float*)d_in[9];
    const float* eb   = (const float*)d_in[10];
    const float* fw   = (const float*)d_in[11];
    const float* fb   = (const float*)d_in[12];
    const float* in_g = (const float*)d_in[13];
    const float* in_b = (const float*)d_in[14];
    const float* dw_w = (const float*)d_in[15];
    const float* dw_b = (const float*)d_in[16];
    const float* n_g  = (const float*)d_in[17];
    const float* n_b  = (const float*)d_in[18];
    const float* p1w  = (const float*)d_in[19];
    const float* p1b  = (const float*)d_in[20];
    const float* p2w  = (const float*)d_in[21];
    const float* p2b  = (const float*)d_in[22];
    const float* outg = (const float*)d_in[23];
    const float* outb = (const float*)d_in[24];
    const float* aw   = (const float*)d_in[25];
    const float* ab   = (const float*)d_in[26];
    const float* kw   = (const float*)d_in[27];
    const float* kb   = (const float*)d_in[28];
    float* out = (float*)d_out;

    char* wp = (char*)d_ws;
    auto take = [&](size_t bytes) -> char* {
        char* r = wp;
        wp += (bytes + 255) & ~(size_t)255;
        return r;
    };

    _Float16* xh     = (_Float16*)take((size_t)4000 * 192 * 2);
    _Float16* cwh    = (_Float16*)take((size_t)512 * 192 * 2);
    _Float16* p1wh   = (_Float16*)take((size_t)6 * 1536 * 512 * 2);
    _Float16* p2wh   = (_Float16*)take((size_t)6 * 512 * 1536 * 2);
    _Float16* kwh    = (_Float16*)take((size_t)1936 * 512 * 2);
    float*    lf0    = (float*)take(4000 * 4);
    float*    en     = (float*)take(4000 * 4);
    float*    ampv   = (float*)take(4000 * 4);
    float*    spkb   = (float*)take(8 * 512 * 4);
    float*    hA     = (float*)take((size_t)4000 * 512 * 4);
    float*    hB     = (float*)take((size_t)4000 * 512 * 4);
    _Float16* hh     = (_Float16*)take((size_t)4000 * 512 * 2);
    _Float16* mid    = (_Float16*)take((size_t)4000 * 1536 * 2);
    float*    Ker    = (float*)take((size_t)4000 * 1936 * 4);
    _Float16* Ftab   = (_Float16*)take((size_t)3872 * 3840 * 2);
    _Float16* Itab   = (_Float16*)take((size_t)3840 * 3872 * 2);
    _Float16* frames = (_Float16*)take((size_t)4008 * 3840 * 2);
    _Float16* Spec   = (_Float16*)take((size_t)4008 * 3872 * 2);
    _Float16* ytime  = (_Float16*)take((size_t)4008 * 3840 * 2);
    float*    harm   = (float*)take((size_t)8 * 480000 * 4);
    float*    Ssum   = (float*)take(8 * 500 * 4);
    double*   Oph    = (double*)take(8 * 500 * 8);

    // --- weight/activation conversions to f16 ---
    cvt_f32_f16_pad<<<(512 * 192 + 255) / 256, 256, 0, stream>>>(cw, cwh, 512, 512, 192);
    cvt_f32_f16_pad<<<(6 * 1536 * 512 + 255) / 256, 256, 0, stream>>>(p1w, p1wh, 6 * 1536, 6 * 1536, 512);
    cvt_f32_f16_pad<<<(6 * 512 * 1536 + 255) / 256, 256, 0, stream>>>(p2w, p2wh, 6 * 512, 6 * 512, 1536);
    cvt_f32_f16_pad<<<(1936 * 512 + 255) / 256, 256, 0, stream>>>(kw, kwh, 1936, 1921, 512);
    xpose_x_f16<<<(4000 * 192 + 255) / 256, 256, 0, stream>>>(x, xh);
    prep_scalars<<<16, 256, 0, stream>>>(f0, energy, lf0, en);
    spk_proj<<<8, 512, 0, stream>>>(spk, sw, cb, sb, fb, eb, spkb);

    // --- input projection (WMMA GEMM) + fused-extras LayerNorm ---
    gemm_wmma_f16<EPI_F32><<<dim3(4, 63), 256, 0, stream>>>(
        xh, cwh, nullptr, nullptr, hB, nullptr, 4000, 512, 192, 512);
    layernorm_c<<<4000, 256, 0, stream>>>(hB, hA, nullptr, in_g, in_b, spkb, fw, ew, lf0, en);

    // --- 6 ConvNeXt blocks ---
    for (int i = 0; i < 6; ++i) {
        dwconv7<<<8000, 256, 0, stream>>>(hA, hB, dw_w + (size_t)i * 512 * 7, dw_b + i * 512);
        layernorm_c<<<4000, 256, 0, stream>>>(hB, nullptr, hh, n_g + i * 512, n_b + i * 512,
                                              nullptr, nullptr, nullptr, nullptr, nullptr);
        gemm_wmma_f16<EPI_GELU_F16><<<dim3(12, 63), 256, 0, stream>>>(
            hh, p1wh + (size_t)i * 1536 * 512, p1b + i * 1536, nullptr,
            nullptr, mid, 4000, 1536, 512, 1536);
        gemm_wmma_f16<EPI_RES_F32><<<dim3(4, 63), 256, 0, stream>>>(
            mid, p2wh + (size_t)i * 512 * 1536, p2b + i * 512, hA,
            hA, nullptr, 4000, 512, 1536, 512);
    }

    // --- output LN + heads ---
    layernorm_c<<<4000, 256, 0, stream>>>(hA, hA, hh, outg, outb,
                                          nullptr, nullptr, nullptr, nullptr, nullptr);
    amps_head<<<500, 256, 0, stream>>>(hA, aw, ab, ampv);
    gemm_wmma_f16<EPI_ELU1_F32><<<dim3(16, 63), 256, 0, stream>>>(
        hh, kwh, kb, nullptr, Ker, nullptr, 4000, 1936, 512, 1921);

    // --- harmonic source ---
    frame_phase_sum<<<dim3(500, 8), 256, 0, stream>>>(f0, Ssum);
    scan_frames<<<8, 512, 0, stream>>>(Ssum, Oph);
    oscillator<<<dim3(500, 8), 32, 0, stream>>>(f0, Oph, ampv, harm);

    // --- STFT noise path: DFT-as-GEMM forward, filter, inverse, OLA ---
    build_ftab<<<3872, 256, 0, stream>>>(Ftab);
    build_itab<<<3840, 256, 0, stream>>>(Itab);
    make_frames<<<dim3(501, 8), 256, 0, stream>>>(noise, frames);
    gemm_wmma_f16<EPI_F16><<<dim3(31, 63), 256, 0, stream>>>(
        frames, Ftab, nullptr, nullptr, nullptr, Spec, 4008, 3872, 3840, 3872);
    apply_kernels<<<60621, 256, 0, stream>>>(Spec, Ker);
    gemm_wmma_f16<EPI_F16><<<dim3(30, 63), 256, 0, stream>>>(
        Spec, Itab, nullptr, nullptr, nullptr, ytime, 4008, 3840, 3872, 3840);
    ola_final<<<dim3(1875, 8), 256, 0, stream>>>(ytime, harm, out);
}
